// EncoderLayer_8873402433762
// MI455X (gfx1250) — matrix operand order repaired and checked by reading, not compiled, not run on hardware
//
#include <hip/hip_runtime.h>
#include <stdint.h>

// ---------------------------------------------------------------------------
// MI455X (gfx1250, wave32) transformer encoder layer.
// Matmuls on V_WMMA_F32_16X16X32_BF16; tile staging via Tensor Data Mover
// (tensor_load_to_lds, TENSORcnt), fp32 accumulation everywhere.
// ---------------------------------------------------------------------------

typedef __attribute__((ext_vector_type(16))) __bf16   v16bf;
typedef __attribute__((ext_vector_type(8)))  float    v8f;
typedef __attribute__((ext_vector_type(4)))  uint32_t u32x4_t;
typedef __attribute__((ext_vector_type(2)))  uint32_t u32x2_t;
typedef __attribute__((ext_vector_type(4)))  int      i32x4_t;
typedef __attribute__((ext_vector_type(8)))  int      i32x8_t;

__device__ __forceinline__ v8f wmma_bf16(v16bf a, v16bf b, v8f c) {
  return __builtin_amdgcn_wmma_f32_16x16x32_bf16(
      false, a, false, b, (short)0, c, false, false);
}

// LDS byte offset of a shared-memory pointer: for the LDS aperture the low
// 32 bits of the flat address are the LDS offset (ISA 10.2 aperture rules).
__device__ __forceinline__ uint32_t lds_off(const void* p) {
  return (uint32_t)(uintptr_t)p;
}

// ---------------------------------------------------------------------------
// Tensor Data Mover: DMA a 2-D tile (tile_d0 elems x tile_d1 rows of bf16)
// from global (row stride stride0 elems) into LDS at lds_byte, inserting
// pad_amt (+1) DWORDs of padding after every 2^(pad_itv+1) DWORDs so the LDS
// pitch matches the bank-conflict-avoiding fragment-reader pitches.
// D# layout per CDNA5 ISA 8.3/8.4.  Issue from ONE wave only.
// ---------------------------------------------------------------------------
__device__ __forceinline__ void tdm_load_2d(
    uint32_t lds_byte, uint64_t gaddr,
    uint32_t tensor_d0, uint32_t tensor_d1, uint64_t stride0,
    uint32_t tile_d0, uint32_t tile_d1,
    uint32_t pad_itv, uint32_t pad_amt, bool do_pad) {
  u32x4_t g0;
  g0[0] = 1u;                                    // count=1, user mode
  g0[1] = lds_byte;                              // lds_addr
  g0[2] = (uint32_t)gaddr;                       // global_addr[31:0]
  g0[3] = (uint32_t)(gaddr >> 32) | (2u << 30);  // addr[56:32] | type=2
  i32x8_t g1;
  uint32_t cfg = (1u << 16);                     // data_size = 2 bytes
  if (do_pad) cfg |= (1u << 20) | (pad_itv << 22) | (pad_amt << 25);
  g1[0] = (int)cfg;
  g1[1] = (int)((tensor_d0 & 0xFFFFu) << 16);            // dim0[15:0]
  g1[2] = (int)((tensor_d0 >> 16) | ((tensor_d1 & 0xFFFFu) << 16));
  g1[3] = (int)((tensor_d1 >> 16) | (tile_d0 << 16));
  g1[4] = (int)(tile_d1 & 0xFFFFu);                      // tile_dim1 (dim2=0)
  g1[5] = (int)(uint32_t)stride0;                        // stride0[31:0]
  g1[6] = (int)(uint32_t)(stride0 >> 32);                // stride0[47:32]
  g1[7] = 0;
  i32x4_t gz = {0, 0, 0, 0};
#if defined(__clang_major__) && __clang_major__ >= 23
  i32x8_t gz8 = {0, 0, 0, 0, 0, 0, 0, 0};
  __builtin_amdgcn_tensor_load_to_lds(g0, g1, gz, gz, gz8, 0);
#else
  __builtin_amdgcn_tensor_load_to_lds(g0, g1, gz, gz, 0);
#endif
}

// --- Fragment loaders (ISA 7.12.2 layouts) ---------------------------------
__device__ __forceinline__ v16bf load_frag_a_g(const __bf16* tile, int stride, int lane) {
  const __bf16* p = tile + (size_t)(lane & 15) * stride + (lane >> 4) * 8;
  union { u32x4_t q[2]; v16bf v; } f;
  f.q[0] = *(const u32x4_t*)(p);
  f.q[1] = *(const u32x4_t*)(p + 16);
  return f.v;
}

__device__ __forceinline__ v16bf load_frag_a_lds(const __bf16* tile, int pitch, int lane) {
  const __bf16* p = tile + (lane & 15) * pitch + (lane >> 4) * 8;
  union { u32x2_t d[4]; v16bf v; } f;
  f.d[0] = *(const u32x2_t*)(p);
  f.d[1] = *(const u32x2_t*)(p + 4);
  f.d[2] = *(const u32x2_t*)(p + 16);
  f.d[3] = *(const u32x2_t*)(p + 20);
  return f.v;
}

__device__ __forceinline__ v16bf load_frag_b(const __bf16* tile, int pitch, int lane) {
  const __bf16* p = tile + (lane & 15) * pitch + (lane >> 4) * 8;
  union { u32x4_t q[2]; v16bf v; } f;
  f.q[0] = *(const u32x4_t*)(p);
  f.q[1] = *(const u32x4_t*)(p + 16);
  return f.v;
}

// ---------------------------------------------------------------------------
__global__ __launch_bounds__(256) void cvt_bf16_kernel(
    const float* __restrict__ in, __bf16* __restrict__ out, int n) {
  int i = blockIdx.x * 256 + threadIdx.x;
  if (i < n) out[i] = (__bf16)in[i];
}

// fp32 [R,C] -> bf16 [C,R] (weights to N-major)
__global__ __launch_bounds__(256) void transpose_cvt_kernel(
    const float* __restrict__ in, __bf16* __restrict__ out, int R, int C) {
  __shared__ __bf16 tile[32][33];
  int c0 = blockIdx.x * 32, r0 = blockIdx.y * 32;
  int lx = threadIdx.x & 31, ly = threadIdx.x >> 5;
  for (int j = 0; j < 4; ++j)
    tile[ly + j * 8][lx] = (__bf16)in[(size_t)(r0 + ly + j * 8) * C + c0 + lx];
  __syncthreads();
  for (int j = 0; j < 4; ++j)
    out[(size_t)(c0 + ly + j * 8) * R + r0 + lx] = tile[lx][ly + j * 8];
}

// V [B,S,H,64] bf16 -> VT [B,H,64,S] bf16
__global__ __launch_bounds__(256) void transpose_head_kernel(
    const __bf16* __restrict__ v, __bf16* __restrict__ vT) {
  constexpr int S = 2048, D = 1024, DH = 64;
  __shared__ __bf16 tile[32][33];
  int bh = blockIdx.z;
  int b = bh >> 4, h = bh & 15;
  int s0 = blockIdx.x * 32, d0 = blockIdx.y * 32;
  int lx = threadIdx.x & 31, ly = threadIdx.x >> 5;
  const __bf16* src = v + ((size_t)b * S) * D + h * DH;
  for (int j = 0; j < 4; ++j)
    tile[ly + j * 8][lx] = src[(size_t)(s0 + ly + j * 8) * D + d0 + lx];
  __syncthreads();
  __bf16* dst = vT + ((size_t)bh * DH) * S;
  for (int j = 0; j < 4; ++j)
    dst[(size_t)(d0 + ly + j * 8) * S + s0 + lx] = tile[lx][ly + j * 8];
}

// ---------------------------------------------------------------------------
// GEMM: C[M,N] = A[M,K](bf16) * BT[N,K](bf16)^T + bias, opt ReLU.
// Block 128x128, BK=32, 8 waves x (32x64).  Tiles staged by the TDM into
// double-buffered LDS; pad cfg (itv=3 -> every 64B, amt=7 -> +32B) gives a
// 48-elem pitch so all ds_load_b128 fragment reads stay 16B aligned.
// ---------------------------------------------------------------------------
template <bool RELU, bool OUT_BF16>
__global__ __launch_bounds__(256, 2) void gemm_kernel(
    const __bf16* __restrict__ A, const __bf16* __restrict__ BT,
    const float* __restrict__ bias, void* __restrict__ Cout,
    int M, int N, int K) {
  __shared__ __bf16 sA[2][128 * 48];
  __shared__ __bf16 sB[2][128 * 48];

  int tid = threadIdx.x, lane = tid & 31, w = tid >> 5;
  int wm = w & 3, wn = w >> 2;
  int m0 = blockIdx.y * 128, n0 = blockIdx.x * 128;
  bool issuer = (tid < 32);  // TDM descriptors issued by wave 0 only

  uint64_t gA64 = (uint64_t)(uintptr_t)A + (uint64_t)m0 * K * 2;
  uint64_t gB64 = (uint64_t)(uintptr_t)BT + (uint64_t)n0 * K * 2;
  uint32_t lA[2] = {lds_off(&sA[0][0]), lds_off(&sA[1][0])};
  uint32_t lB[2] = {lds_off(&sB[0][0]), lds_off(&sB[1][0])};

  v8f acc[2][4];
  for (int mi = 0; mi < 2; ++mi)
    for (int ni = 0; ni < 4; ++ni) acc[mi][ni] = (v8f){};

  int nk = K >> 5;
  if (issuer) {
    tdm_load_2d(lA[0], gA64, K, M, K, 32, 128, 3, 7, true);
    tdm_load_2d(lB[0], gB64, K, N, K, 32, 128, 3, 7, true);
    __builtin_amdgcn_s_wait_tensorcnt(0);
  }
  __syncthreads();

  for (int kt = 0; kt < nk; ++kt) {
    int buf = kt & 1;
    if (issuer && kt + 1 < nk) {
      uint64_t koff = (uint64_t)(kt + 1) * 64;  // 32 elems * 2B
      tdm_load_2d(lA[buf ^ 1], gA64 + koff, K, M, K, 32, 128, 3, 7, true);
      tdm_load_2d(lB[buf ^ 1], gB64 + koff, K, N, K, 32, 128, 3, 7, true);
    }

    const __bf16* tA = sA[buf];
    const __bf16* tB = sB[buf];
    v16bf af[2], bfv[4];
    for (int mi = 0; mi < 2; ++mi)
      af[mi] = load_frag_a_g(tA + (wm * 32 + mi * 16) * 48, 48, lane);
    for (int ni = 0; ni < 4; ++ni)
      bfv[ni] = load_frag_b(tB + (wn * 64 + ni * 16) * 48, 48, lane);
    for (int mi = 0; mi < 2; ++mi)
      for (int ni = 0; ni < 4; ++ni)
        acc[mi][ni] = wmma_bf16(af[mi], bfv[ni], acc[mi][ni]);

    if (issuer) __builtin_amdgcn_s_wait_tensorcnt(0);
    __syncthreads();
  }

  int hi = lane >> 4, c16 = lane & 15;
  for (int mi = 0; mi < 2; ++mi)
    for (int ni = 0; ni < 4; ++ni) {
      int col = n0 + wn * 64 + ni * 16 + c16;
      int row0 = m0 + wm * 32 + mi * 16 + 8 * hi;
      float bv = bias ? bias[col] : 0.f;
      for (int r = 0; r < 8; ++r) {
        float val = acc[mi][ni][r] + bv;
        if (RELU) val = fmaxf(val, 0.f);
        if (OUT_BF16)
          ((__bf16*)Cout)[(size_t)(row0 + r) * N + col] = (__bf16)val;
        else
          ((float*)Cout)[(size_t)(row0 + r) * N + col] = val;
      }
    }
}

// ---------------------------------------------------------------------------
// Attention, one-pass streaming softmax (logits clipped to +-10 before
// softmax => exp bounded, no running-max rescale).  K/V tiles staged via TDM
// (pitches 72 / 48 via pad cfg); row sums from a WMMA against an all-ones B
// operand so the final divide stays in-lane.
// ---------------------------------------------------------------------------
__global__ __launch_bounds__(128) void attn_kernel(
    const __bf16* __restrict__ Q, const __bf16* __restrict__ Kb,
    const __bf16* __restrict__ VT, __bf16* __restrict__ Ctx) {
  constexpr int S = 2048, D = 1024, DH = 64;
  __shared__ __bf16 sK[32 * 72];     // 32 keys x 64 dh, pitch 72 (128B+16B)
  __shared__ __bf16 sV[64 * 48];     // 64 dh x 32 keys, pitch 48 (64B+32B)
  __shared__ __bf16 sP[4][16 * 40];  // per-wave exp(P) relayout tile

  int tid = threadIdx.x, lane = tid & 31, wv = tid >> 5;
  int bh = blockIdx.y, b = bh >> 4, h = bh & 15;
  int q0 = blockIdx.x * 64 + wv * 16;

  const __bf16* qbase = Q + ((size_t)(b * S + q0)) * D + h * DH;
  v16bf qf0 = load_frag_a_g(qbase, D, lane);
  v16bf qf1 = load_frag_a_g(qbase + 32, D, lane);

  v16bf ones;
  for (int i = 0; i < 16; ++i) ones[i] = (__bf16)1.0f;

  v8f cacc[4]; v8f sacc = (v8f){};
  for (int ni = 0; ni < 4; ++ni) cacc[ni] = (v8f){};

  uint64_t kbase = (uint64_t)(uintptr_t)Kb + ((uint64_t)b * S * D + h * DH) * 2;
  uint64_t vbase = (uint64_t)(uintptr_t)VT + ((uint64_t)bh * DH * S) * 2;
  uint32_t lK = lds_off(sK), lV = lds_off(sV);
  int hi = lane >> 4, col = lane & 15;
  __bf16* pw = sP[wv];

  for (int kk = 0; kk < S; kk += 32) {
    __syncthreads();
    if (tid < 32) {
      // K tile: 64 elems x 32 key-rows, row stride D; pad 128B->+16B (itv 4, amt 3)
      tdm_load_2d(lK, kbase + (uint64_t)kk * D * 2, 64, S, D, 64, 32, 4, 3, true);
      // VT tile: 32 elems x 64 dh-rows, row stride S; pad 64B->+32B (itv 3, amt 7)
      tdm_load_2d(lV, vbase + (uint64_t)kk * 2, 32, DH * S, S, 32, 64, 3, 7, true);
      __builtin_amdgcn_s_wait_tensorcnt(0);
    }
    __syncthreads();

    v8f l0 = (v8f){}, l1 = (v8f){};
    l0 = wmma_bf16(qf0, load_frag_b(sK, 72, lane), l0);
    l0 = wmma_bf16(qf1, load_frag_b(sK + 32, 72, lane), l0);
    l1 = wmma_bf16(qf0, load_frag_b(sK + 16 * 72, 72, lane), l1);
    l1 = wmma_bf16(qf1, load_frag_b(sK + 16 * 72 + 32, 72, lane), l1);

    for (int r = 0; r < 8; ++r) {
      float e0 = __expf(fminf(fmaxf(l0[r] * 0.125f, -10.f), 10.f));
      float e1 = __expf(fminf(fmaxf(l1[r] * 0.125f, -10.f), 10.f));
      pw[(r + 8 * hi) * 40 + col]      = (__bf16)e0;
      pw[(r + 8 * hi) * 40 + col + 16] = (__bf16)e1;
    }
    v16bf pf = load_frag_a_lds(pw, 40, lane);  // same-wave LDS, DScnt ordered

    sacc = wmma_bf16(pf, ones, sacc);
    for (int ni = 0; ni < 4; ++ni)
      cacc[ni] = wmma_bf16(pf, load_frag_b(sV + ni * 16 * 48, 48, lane), cacc[ni]);
  }

  __bf16* obase = Ctx + ((size_t)(b * S + q0)) * D + h * DH;
  for (int ni = 0; ni < 4; ++ni)
    for (int r = 0; r < 8; ++r) {
      float val = cacc[ni][r] / sacc[r];
      obase[(size_t)(r + 8 * hi) * D + ni * 16 + col] = (__bf16)val;
    }
}

// ---------------------------------------------------------------------------
// out = LayerNorm(A + B); optionally also emit bf16 copy for next GEMM.
// ---------------------------------------------------------------------------
__global__ __launch_bounds__(256) void add_ln_kernel(
    const float* __restrict__ A, const float* __restrict__ B,
    const float* __restrict__ gamma, const float* __restrict__ beta,
    float* __restrict__ outf, __bf16* __restrict__ outh, int d) {
  int row = blockIdx.x;
  const float* a = A + (size_t)row * d;
  const float* b = B + (size_t)row * d;
  float v[4], s = 0.f, ss = 0.f;
  for (int j = 0; j < 4; ++j) {
    int c = threadIdx.x + j * 256;
    float t = a[c] + b[c];
    v[j] = t; s += t; ss += t * t;
  }
  int lane = threadIdx.x & 31, wid = threadIdx.x >> 5;
  for (int o = 16; o > 0; o >>= 1) {
    s += __shfl_xor(s, o);
    ss += __shfl_xor(ss, o);
  }
  __shared__ float red[16];
  if (lane == 0) { red[wid] = s; red[8 + wid] = ss; }
  __syncthreads();
  float ts = 0.f, tss = 0.f;
  for (int i = 0; i < 8; ++i) { ts += red[i]; tss += red[8 + i]; }
  float mu = ts / d;
  float var = tss / d - mu * mu;
  float rstd = rsqrtf(var + 1e-8f);
  for (int j = 0; j < 4; ++j) {
    int c = threadIdx.x + j * 256;
    float o = (v[j] - mu) * rstd * gamma[c] + beta[c];
    outf[(size_t)row * d + c] = o;
    if (outh) outh[(size_t)row * d + c] = (__bf16)o;
  }
}

// ---------------------------------------------------------------------------
extern "C" void kernel_launch(void* const* d_in, const int* in_sizes, int n_in,
                              void* d_out, int out_size, void* d_ws, size_t ws_size,
                              hipStream_t stream) {
  constexpr int Btch = 4, S = 2048, D = 1024, H = 16, F = 4096;
  constexpr size_t TOK = (size_t)Btch * S;

  const float* x     = (const float*)d_in[0];
  const float* wq    = (const float*)d_in[1];
  const float* bq    = (const float*)d_in[2];
  const float* wk    = (const float*)d_in[3];
  const float* bk    = (const float*)d_in[4];
  const float* wvp   = (const float*)d_in[5];
  const float* bv    = (const float*)d_in[6];
  const float* wo    = (const float*)d_in[7];
  const float* bo    = (const float*)d_in[8];
  const float* w1    = (const float*)d_in[9];
  const float* b1    = (const float*)d_in[10];
  const float* w2    = (const float*)d_in[11];
  const float* b2    = (const float*)d_in[12];
  const float* gamma = (const float*)d_in[13];
  const float* beta  = (const float*)d_in[14];

  char* ws = (char*)d_ws;
  size_t off = 0;
  auto alloc = [&](size_t bytes) {
    size_t r = off;
    off = (off + bytes + 255) & ~(size_t)255;
    return r;
  };
  size_t o_xh  = alloc(TOK * D * 2);
  size_t o_wqT = alloc((size_t)D * D * 2);
  size_t o_wkT = alloc((size_t)D * D * 2);
  size_t o_wvT = alloc((size_t)D * D * 2);
  size_t o_woT = alloc((size_t)D * D * 2);
  size_t o_w1T = alloc((size_t)F * D * 2);
  size_t o_w2T = alloc((size_t)D * F * 2);
  size_t o_q   = alloc(TOK * D * 2);
  size_t o_k   = alloc(TOK * D * 2);   // k..ctx (64MB contiguous) reused as hid
  size_t o_v   = alloc(TOK * D * 2);
  size_t o_vT  = alloc(TOK * D * 2);
  size_t o_ctx = alloc(TOK * D * 2);
  size_t o_tmp = alloc(TOK * D * 4);
  size_t o_out1= alloc(TOK * D * 4);
  size_t o_o1h = o_q;                  // reuse: q dead after attention
  size_t o_hid = o_k;                  // reuse: k..ctx dead after LN1

  __bf16* xh  = (__bf16*)(ws + o_xh);
  __bf16* wqT = (__bf16*)(ws + o_wqT);
  __bf16* wkT = (__bf16*)(ws + o_wkT);
  __bf16* wvT = (__bf16*)(ws + o_wvT);
  __bf16* woT = (__bf16*)(ws + o_woT);
  __bf16* w1T = (__bf16*)(ws + o_w1T);
  __bf16* w2T = (__bf16*)(ws + o_w2T);
  __bf16* qb  = (__bf16*)(ws + o_q);
  __bf16* kb  = (__bf16*)(ws + o_k);
  __bf16* vb  = (__bf16*)(ws + o_v);
  __bf16* vTb = (__bf16*)(ws + o_vT);
  __bf16* ctx = (__bf16*)(ws + o_ctx);
  float*  tmp = (float*)(ws + o_tmp);
  float*  out1= (float*)(ws + o_out1);
  __bf16* o1h = (__bf16*)(ws + o_o1h);
  __bf16* hid = (__bf16*)(ws + o_hid);

  cvt_bf16_kernel<<<(int)(TOK * D / 256), 256, 0, stream>>>(x, xh, (int)(TOK * D));
  transpose_cvt_kernel<<<dim3(D / 32, D / 32), 256, 0, stream>>>(wq, wqT, D, D);
  transpose_cvt_kernel<<<dim3(D / 32, D / 32), 256, 0, stream>>>(wk, wkT, D, D);
  transpose_cvt_kernel<<<dim3(D / 32, D / 32), 256, 0, stream>>>(wvp, wvT, D, D);
  transpose_cvt_kernel<<<dim3(D / 32, D / 32), 256, 0, stream>>>(wo, woT, D, D);
  transpose_cvt_kernel<<<dim3(F / 32, D / 32), 256, 0, stream>>>(w1, w1T, D, F);
  transpose_cvt_kernel<<<dim3(D / 32, F / 32), 256, 0, stream>>>(w2, w2T, F, D);

  gemm_kernel<false, true><<<dim3(D / 128, (int)(TOK / 128)), 256, 0, stream>>>(
      xh, wqT, bq, qb, (int)TOK, D, D);
  gemm_kernel<false, true><<<dim3(D / 128, (int)(TOK / 128)), 256, 0, stream>>>(
      xh, wkT, bk, kb, (int)TOK, D, D);
  gemm_kernel<false, true><<<dim3(D / 128, (int)(TOK / 128)), 256, 0, stream>>>(
      xh, wvT, bv, vb, (int)TOK, D, D);

  transpose_head_kernel<<<dim3(S / 32, 2, Btch * H), 256, 0, stream>>>(vb, vTb);
  attn_kernel<<<dim3(S / 64, Btch * H), 128, 0, stream>>>(qb, kb, vTb, ctx);

  gemm_kernel<false, false><<<dim3(D / 128, (int)(TOK / 128)), 256, 0, stream>>>(
      ctx, woT, bo, tmp, (int)TOK, D, D);
  add_ln_kernel<<<(int)TOK, 256, 0, stream>>>(x, tmp, gamma, beta, out1, o1h, D);

  gemm_kernel<true, true><<<dim3(F / 128, (int)(TOK / 128)), 256, 0, stream>>>(
      o1h, w1T, b1, hid, (int)TOK, F, D);
  gemm_kernel<false, false><<<dim3(D / 128, (int)(TOK / 128)), 256, 0, stream>>>(
      hid, w2T, b2, tmp, (int)TOK, D, F);
  add_ln_kernel<<<(int)TOK, 256, 0, stream>>>(out1, tmp, gamma, beta,
                                              (float*)d_out, (__bf16*)nullptr, D);
}
